// NaiveMultiHeadAttention_65549790871657
// MI455X (gfx1250) — matrix operand order repaired and checked by reading, not compiled, not run on hardware
//
#include <hip/hip_runtime.h>
#include <hip/hip_bf16.h>
#include <stdint.h>

typedef __bf16 bf16;
typedef __attribute__((ext_vector_type(16))) __bf16 v16bf;
typedef __attribute__((ext_vector_type(8)))  float   v8f;
typedef __attribute__((ext_vector_type(8)))  __bf16  bf16x8;
typedef __attribute__((ext_vector_type(4)))  __bf16  bf16x4;
typedef __attribute__((ext_vector_type(4)))  float   f32x4;

union Frag { v16bf v; bf16x8 h[2]; };

#define EXP2F(x) __builtin_amdgcn_exp2f(x)   // v_exp_f32 (hardware exp2)

// CDNA5 async DMA: global -> LDS, 16B per lane, tracked by ASYNCcnt.
// Low 32 bits of a generic LDS pointer are the wave-relative LDS offset
// (flat-aperture model, ISA 10.2), which is what VDST expects.
__device__ __forceinline__ void async_ld_b128(bf16* lds, const bf16* gaddr) {
  asm volatile("global_load_async_to_lds_b128 %0, %1, off"
               :: "v"((uint32_t)(uintptr_t)lds), "v"(gaddr)
               : "memory");
}
__device__ __forceinline__ void wait_async0() {
  asm volatile("s_wait_asynccnt 0x0" ::: "memory");
}

// ---------------------------------------------------------------------------
// f32 -> bf16 (RNE), vectorized 4-wide
// ---------------------------------------------------------------------------
__global__ __launch_bounds__(256) void cvt_f32_to_bf16(const float* __restrict__ in,
                                                       bf16* __restrict__ out, int n4) {
  int i = blockIdx.x * blockDim.x + threadIdx.x;
  if (i >= n4) return;
  f32x4 v = ((const f32x4*)in)[i];
  bf16x4 o;
  o[0] = (bf16)v[0]; o[1] = (bf16)v[1]; o[2] = (bf16)v[2]; o[3] = (bf16)v[3];
  ((bf16x4*)out)[i] = o;
}

// ---------------------------------------------------------------------------
// Projection GEMM: Y[m,f] = sum_e X[m,e] * W[f,e]   (y = x @ W^T)
// Block tile 128x128, 4 waves, wave tile 64x64, K-chunk 32.
// Double-buffered LDS, staged entirely by async global->LDS DMA.
// ---------------------------------------------------------------------------
__global__ __launch_bounds__(128) void proj_gemm_bf16(const bf16* __restrict__ X,
                                                      const bf16* __restrict__ W,
                                                      bf16* __restrict__ Y) {
  __shared__ bf16 lA[2][128][40];   // [buf][row][32K + 8 pad] (16B-aligned cols)
  __shared__ bf16 lB[2][128][40];

  const int f0   = blockIdx.x * 128;
  const int m0   = blockIdx.y * 128;
  const int tid  = threadIdx.x;
  const int lane = tid & 31;
  const int wid  = tid >> 5;
  const int wm   = (wid >> 1) * 64;
  const int wn   = (wid & 1) * 64;
  const int l16  = lane & 15;
  const int hi   = lane >> 4;

  v8f acc[4][4] = {};

  auto stage = [&](int buf, int kk) {
#pragma unroll
    for (int it = 0; it < 4; ++it) {          // 512 chunks / 128 threads
      int c = tid + it * 128;
      int row = c >> 2, col = (c & 3) * 8;
      async_ld_b128(&lA[buf][row][col], &X[(size_t)(m0 + row) * 1024 + kk + col]);
      async_ld_b128(&lB[buf][row][col], &W[(size_t)(f0 + row) * 1024 + kk + col]);
    }
  };

  stage(0, 0);
  wait_async0();
  __syncthreads();

  for (int kk = 0; kk < 1024; kk += 32) {
    const int cur = (kk >> 5) & 1;
    if (kk + 32 < 1024) stage(cur ^ 1, kk + 32);   // overlap DMA with WMMA

    Frag a[4], b[4];
#pragma unroll
    for (int t = 0; t < 4; ++t) {   // A 16x32: lane<16 K0..7,16..23; lane>=16 K8..15,24..31
      int row = wm + t * 16 + l16;
      a[t].h[0] = *(const bf16x8*)&lA[cur][row][hi * 8];
      a[t].h[1] = *(const bf16x8*)&lA[cur][row][16 + hi * 8];
    }
#pragma unroll
    for (int n = 0; n < 4; ++n) {   // B 32x16: lane = column f, lane<16 K0..7,16..23; lane>=16 K8..15,24..31
      int frow = wn + n * 16 + l16;
      b[n].h[0] = *(const bf16x8*)&lB[cur][frow][hi * 8];
      b[n].h[1] = *(const bf16x8*)&lB[cur][frow][16 + hi * 8];
    }
#pragma unroll
    for (int t = 0; t < 4; ++t)
#pragma unroll
      for (int n = 0; n < 4; ++n)
        acc[t][n] = __builtin_amdgcn_wmma_f32_16x16x32_bf16(
            false, a[t].v, false, b[n].v, (short)0, acc[t][n], false, false);

    wait_async0();     // my DMA into buf^1 landed
    __syncthreads();   // everyone's DMA landed; everyone's reads of cur done
  }

#pragma unroll
  for (int t = 0; t < 4; ++t)
#pragma unroll
    for (int n = 0; n < 4; ++n)
#pragma unroll
      for (int r = 0; r < 8; ++r) {
        int m = m0 + wm + t * 16 + r + hi * 8;   // C layout: reg r -> M = r + hi*8
        int f = f0 + wn + n * 16 + l16;          //           lane  -> N = l16
        Y[(size_t)m * 1024 + f] = (bf16)acc[t][n][r];
      }
}

// ---------------------------------------------------------------------------
// Flash attention over 64 groups of [1024 x 64] bf16 (contiguous slices).
// Block = 4 waves, owns (group, 64-row Q tile); wave owns 16 rows.
// Q/K staged by async DMA; softmax in base-2 with 1/8*log2(e) folded in.
// ---------------------------------------------------------------------------
__global__ __launch_bounds__(128) void mha_flash(const bf16* __restrict__ Q,
                                                 const bf16* __restrict__ K,
                                                 const bf16* __restrict__ V,
                                                 float* __restrict__ out) {
  __shared__ bf16 sQ[64][72];
  __shared__ bf16 sK[64][72];
  __shared__ bf16 sVt[64][72];      // transposed: [d][key]
  __shared__ bf16 sP[4][16][72];    // per-wave exp2 tile

  const float SCL = 0.125f * 1.44269504088896f;  // 1/sqrt(64) * log2(e)

  const int g     = blockIdx.y;
  const int qbase = blockIdx.x * 64;
  const int tid   = threadIdx.x;
  const int lane  = tid & 31;
  const int w     = tid >> 5;
  const int l16   = lane & 15;
  const int hi    = lane >> 4;

  const bf16* Qg = Q + (size_t)g * 65536;   // group = contiguous [1024][64] slice
  const bf16* Kg = K + (size_t)g * 65536;
  const bf16* Vg = V + (size_t)g * 65536;

  // stage Q tile via async DMA
#pragma unroll
  for (int it = 0; it < 4; ++it) {
    int c = tid + it * 128;
    int row = c >> 3, col = (c & 7) * 8;
    async_ld_b128(&sQ[row][col], &Qg[(size_t)(qbase + row) * 64 + col]);
  }
  wait_async0();
  __syncthreads();

  Frag qa[2];
#pragma unroll
  for (int ks = 0; ks < 2; ++ks) {
    int row = w * 16 + l16;
    qa[ks].h[0] = *(const bf16x8*)&sQ[row][ks * 32 + hi * 8];
    qa[ks].h[1] = *(const bf16x8*)&sQ[row][ks * 32 + 16 + hi * 8];
  }

  float m_r[8], l_r[8];
  v8f o[4] = {};
#pragma unroll
  for (int r = 0; r < 8; ++r) { m_r[r] = -1e30f; l_r[r] = 0.f; }

  for (int j = 0; j < 16; ++j) {
    const int kb = j * 64;
    __syncthreads();                 // previous tile's readers done
#pragma unroll
    for (int it = 0; it < 4; ++it) {
      int c = tid + it * 128;
      int row = c >> 3, col = (c & 7) * 8;
      async_ld_b128(&sK[row][col], &Kg[(size_t)(kb + row) * 64 + col]);
      bf16x8 vv = *(const bf16x8*)&Vg[(size_t)(kb + row) * 64 + col];
#pragma unroll
      for (int i = 0; i < 8; ++i) sVt[col + i][row] = vv[i];
    }
    wait_async0();
    __syncthreads();

    // S = Q K^T : A = Q rows, B = K rows along d (= columns of K^T)
    v8f s[4] = {};
#pragma unroll
    for (int ks = 0; ks < 2; ++ks)
#pragma unroll
      for (int n = 0; n < 4; ++n) {
        Frag b;
        int krow = n * 16 + l16;
        b.h[0] = *(const bf16x8*)&sK[krow][ks * 32 + hi * 8];
        b.h[1] = *(const bf16x8*)&sK[krow][ks * 32 + 16 + hi * 8];
        s[n] = __builtin_amdgcn_wmma_f32_16x16x32_bf16(
            false, qa[ks].v, false, b.v, (short)0, s[n], false, false);
      }
#pragma unroll
    for (int n = 0; n < 4; ++n)
#pragma unroll
      for (int r = 0; r < 8; ++r) s[n][r] *= SCL;   // scale into log2 domain

    // flash softmax update (lanes<16 -> M=r, lanes>=16 -> M=r+8)
    float alpha[8];
#pragma unroll
    for (int r = 0; r < 8; ++r) {
      float rm = fmaxf(fmaxf(s[0][r], s[1][r]), fmaxf(s[2][r], s[3][r]));
#pragma unroll
      for (int msk = 1; msk <= 8; msk <<= 1) rm = fmaxf(rm, __shfl_xor(rm, msk, 32));
      float mn = fmaxf(m_r[r], rm);
      alpha[r] = EXP2F(m_r[r] - mn);
      m_r[r] = mn;
    }
#pragma unroll
    for (int n = 0; n < 4; ++n)
#pragma unroll
      for (int r = 0; r < 8; ++r) {
        float e = EXP2F(s[n][r] - m_r[r]);
        s[n][r] = e;
        sP[w][r + hi * 8][n * 16 + l16] = (bf16)e;
      }
#pragma unroll
    for (int r = 0; r < 8; ++r) {
      float rs = s[0][r] + s[1][r] + s[2][r] + s[3][r];
#pragma unroll
      for (int msk = 1; msk <= 8; msk <<= 1) rs += __shfl_xor(rs, msk, 32);
      l_r[r] = l_r[r] * alpha[r] + rs;
#pragma unroll
      for (int n = 0; n < 4; ++n) o[n][r] *= alpha[r];
    }

    // O += P @ V : A = P rows along keys, B = V^T rows (d) along keys
#pragma unroll
    for (int ks = 0; ks < 2; ++ks) {
      Frag pa;
      pa.h[0] = *(const bf16x8*)&sP[w][l16][ks * 32 + hi * 8];
      pa.h[1] = *(const bf16x8*)&sP[w][l16][ks * 32 + 16 + hi * 8];
#pragma unroll
      for (int n = 0; n < 4; ++n) {
        Frag b;
        int drow = n * 16 + l16;
        b.h[0] = *(const bf16x8*)&sVt[drow][ks * 32 + hi * 8];
        b.h[1] = *(const bf16x8*)&sVt[drow][ks * 32 + 16 + hi * 8];
        o[n] = __builtin_amdgcn_wmma_f32_16x16x32_bf16(
            false, pa.v, false, b.v, (short)0, o[n], false, false);
      }
    }
  }

  // epilogue: out is plain contiguous f32 (reshape back is a no-op)
#pragma unroll
  for (int n = 0; n < 4; ++n)
#pragma unroll
    for (int r = 0; r < 8; ++r) {
      int row = qbase + w * 16 + r + hi * 8;
      int d   = n * 16 + l16;
      out[(size_t)g * 65536 + (size_t)row * 64 + d] = o[n][r] / l_r[r];
    }
}

// ---------------------------------------------------------------------------
extern "C" void kernel_launch(void* const* d_in, const int* in_sizes, int n_in,
                              void* d_out, int out_size, void* d_ws, size_t ws_size,
                              hipStream_t stream) {
  const float* q  = (const float*)d_in[0];
  const float* k  = (const float*)d_in[1];
  const float* v  = (const float*)d_in[2];
  const float* Wq = (const float*)d_in[3];
  const float* Wk = (const float*)d_in[4];
  const float* Wv = (const float*)d_in[5];
  float* out = (float*)d_out;

  const size_t XE = (size_t)4096 * 1024;
  const size_t WE = (size_t)1024 * 1024;

  bf16* ws  = (bf16*)d_ws;
  bf16* xq  = ws;        bf16* xk  = xq + XE;   bf16* xv  = xk + XE;
  bf16* wqb = xv + XE;   bf16* wkb = wqb + WE;  bf16* wvb = wkb + WE;
  bf16* Qp  = wvb + WE;  bf16* Kp  = Qp + XE;   bf16* Vp  = Kp + XE;

  const int xb = (int)(XE / 4), wb = (int)(WE / 4);
  cvt_f32_to_bf16<<<(xb + 255) / 256, 256, 0, stream>>>(q,  xq,  xb);
  cvt_f32_to_bf16<<<(xb + 255) / 256, 256, 0, stream>>>(k,  xk,  xb);
  cvt_f32_to_bf16<<<(xb + 255) / 256, 256, 0, stream>>>(v,  xv,  xb);
  cvt_f32_to_bf16<<<(wb + 255) / 256, 256, 0, stream>>>(Wq, wqb, wb);
  cvt_f32_to_bf16<<<(wb + 255) / 256, 256, 0, stream>>>(Wk, wkb, wb);
  cvt_f32_to_bf16<<<(wb + 255) / 256, 256, 0, stream>>>(Wv, wvb, wb);

  dim3 gg(8, 32);   // F tiles x M tiles
  proj_gemm_bf16<<<gg, 128, 0, stream>>>(xq, wqb, Qp);
  proj_gemm_bf16<<<gg, 128, 0, stream>>>(xk, wkb, Kp);
  proj_gemm_bf16<<<gg, 128, 0, stream>>>(xv, wvb, Vp);

  mha_flash<<<dim3(16, 64), 128, 0, stream>>>(Qp, Kp, Vp, out);
}
